// TransformerBlock_54709293416778
// MI455X (gfx1250) — compile-verified, matrix operand order checked by reading, not run on hardware
//
#include <hip/hip_runtime.h>

// ---- problem dims ----
#define B_   2
#define S_   2048
#define D_   1024
#define H_   16
#define DFF_ 4096
#define HD_  64
#define M_   (B_ * S_)   // 4096 rows

typedef unsigned short u16;
typedef unsigned int   u32;
typedef __attribute__((ext_vector_type(4)))  u32    u32x4;
typedef __attribute__((ext_vector_type(8)))  u32    u32x8;
typedef __attribute__((ext_vector_type(16))) __bf16 bf16x16;
typedef __attribute__((ext_vector_type(8)))  float  f32x8;

union FragBF { bf16x16 v; u32x4 u[2]; };
union Pack8  { u32x4 u; u16 h[8]; };

__device__ __forceinline__ u16 f2bf(float f) {
  union { float f; u32 u; } c; c.f = f;
  u32 r = c.u + 0x7fffu + ((c.u >> 16) & 1u);   // round-to-nearest-even
  return (u16)(r >> 16);
}
__device__ __forceinline__ float bf2f(u16 h) {
  union { u32 u; float f; } c; c.u = ((u32)h) << 16;
  return c.f;
}

// flat shared-pointer low 32 bits == LDS byte offset within the workgroup
__device__ __forceinline__ unsigned lds_addr32(const void* p) {
  return (unsigned)(unsigned long long)p;
}

// ---- CDNA5 async global->LDS copy (ASYNCcnt) ----
__device__ __forceinline__ void async_load_b128(const void* gaddr, unsigned lds) {
  asm volatile("global_load_async_to_lds_b128 %0, %1, off"
               :: "v"(lds), "v"(gaddr) : "memory");
}
__device__ __forceinline__ void wait_asynccnt0() {
  asm volatile("s_wait_asynccnt 0x0" ::: "memory");
}

// ---- CDNA5 Tensor Data Mover: 2D tile (rows x row_elems bf16) -> LDS ----
// LDS padding: 4 DWORDs (8 halves) after every 32 DWORDs (64 halves) => row
// stride 72 halves, matching the padded LDS tile layout used by the fragments.
__device__ __forceinline__ void tdm_load_tile_2d(const void* gaddr, unsigned lds,
                                                 unsigned k_elems, unsigned rows,
                                                 unsigned row_elems) {
  const unsigned long long ga = (unsigned long long)gaddr;
  u32x4 g0;
  g0[0] = 1u;                                   // count=1, user mode, no gather
  g0[1] = lds;                                  // lds_addr (bytes)
  g0[2] = (u32)ga;                              // global_addr[31:0] (tile start)
  g0[3] = (u32)(ga >> 32) | (2u << 30);         // global_addr[56:32] | type=2
  u32x8 g1;
  g1[0] = (1u << 16)                            // data_size = 2 bytes
        | (1u << 20)                            // pad_enable
        | (4u << 22)                            // pad_interval: every 32 DWORDs
        | (3u << 25);                           // pad_amount: 4 DWORDs
  g1[1] = (k_elems & 0xFFFFu) << 16;            // tensor_dim0[15:0]
  g1[2] = (k_elems >> 16) & 0xFFFFu;            // tensor_dim0[31:16], dim1[15:0]=0
  g1[3] = 0x10u | (row_elems << 16);            // tensor_dim1=1M rows | tile_dim0
  g1[4] = rows;                                 // tile_dim1 (tile_dim2 = 0)
  g1[5] = k_elems;                              // tensor_dim0_stride[31:0]
  g1[6] = 0u;                                   // stride hi, dim1_stride lo
  g1[7] = 0u;
  asm volatile("tensor_load_to_lds %0, %1" :: "s"(g0), "s"(g1) : "memory");
}

// 16-lane (half-wave) butterfly via ds_swizzle group-of-32 xor mode
template<int MASK>
__device__ __forceinline__ float lane_xor_f(float x) {
  int i = __builtin_amdgcn_ds_swizzle(__float_as_int(x), (MASK << 10) | 0x1f);
  return __int_as_float(i);
}
__device__ __forceinline__ float red16_max(float v) {
  v = fmaxf(v, lane_xor_f<1>(v)); v = fmaxf(v, lane_xor_f<2>(v));
  v = fmaxf(v, lane_xor_f<4>(v)); v = fmaxf(v, lane_xor_f<8>(v));
  return v;
}
__device__ __forceinline__ float red16_add(float v) {
  v += lane_xor_f<1>(v); v += lane_xor_f<2>(v);
  v += lane_xor_f<4>(v); v += lane_xor_f<8>(v);
  return v;
}
__device__ __forceinline__ float red32_add(float v) {
  v = red16_add(v); v += lane_xor_f<16>(v);
  return v;
}

// ---------------------------------------------------------------------------
// fp32 -> bf16 convert (weights)
// ---------------------------------------------------------------------------
__global__ __launch_bounds__(256)
void cvt_f32_bf16(const float* __restrict__ src, u16* __restrict__ dst, long n) {
  const long i = (long)blockIdx.x * 256 + threadIdx.x;
  if (i < n) dst[i] = f2bf(src[i]);
}

// ---------------------------------------------------------------------------
// RMSNorm: one block per row of D_, writes bf16
// ---------------------------------------------------------------------------
__global__ __launch_bounds__(256)
void rmsnorm_bf16(const float* __restrict__ x, const float* __restrict__ g,
                  u16* __restrict__ outb) {
  __shared__ float red[8];
  const long row = blockIdx.x;
  const float* xr = &x[row * (long)D_];
  const int tid = threadIdx.x;
  float vals[4];
  float ss = 0.0f;
#pragma unroll
  for (int i = 0; i < 4; ++i) {
    vals[i] = xr[tid + i * 256];
    ss += vals[i] * vals[i];
  }
  ss = red32_add(ss);
  if ((tid & 31) == 0) red[tid >> 5] = ss;
  __syncthreads();
  float tot = 0.0f;
#pragma unroll
  for (int w = 0; w < 8; ++w) tot += red[w];
  const float rms = rsqrtf(tot * (1.0f / (float)D_) + 1e-6f);
#pragma unroll
  for (int i = 0; i < 4; ++i) {
    const int c = tid + i * 256;
    outb[row * (long)D_ + c] = f2bf(vals[i] * rms * g[c]);
  }
}

// ---------------------------------------------------------------------------
// RoPE, in place on bf16 [B,S,H,HD]; one thread per (row, head, pair)
// ---------------------------------------------------------------------------
__global__ __launch_bounds__(256)
void rope_inplace(u16* __restrict__ t) {
  const long gid = (long)blockIdx.x * 256 + threadIdx.x;
  const long total = (long)M_ * H_ * 32;
  if (gid >= total) return;
  const int  i   = (int)(gid & 31);
  const long rem = gid >> 5;
  const int  h   = (int)(rem & (H_ - 1));
  const long m   = rem >> 4;                 // H_ == 16
  const int  pos = (int)(m & (S_ - 1));
  const float inv = __expf(-(float)i * (9.210340371976184f / 32.0f)); // 1e4^(-i/32)
  const float ang = (float)pos * inv;
  float sn, cs;
  __sincosf(ang, &sn, &cs);
  const long off = m * (long)D_ + (long)h * HD_ + i;
  const float x0 = bf2f(t[off]);
  const float x1 = bf2f(t[off + 32]);
  t[off]      = f2bf(x0 * cs - x1 * sn);
  t[off + 32] = f2bf(x1 * cs + x0 * sn);
}

// ---------------------------------------------------------------------------
// Tiled WMMA GEMM: C[M,N] = A[M,K] (bf16, row-major) x W[N,K]^T (bf16, row-major)
// Block tile 128x128, 8 waves (2x4), wave tile 64x32 = 4x2 WMMA 16x16 tiles.
// A/W tiles DMA'd by the Tensor Data Mover into double-buffered padded LDS;
// wave 0 issues descriptors, pipeline depth 1 tile, s_wait_tensorcnt gates.
// Epilogues: 0 bf16, 1 silu->bf16, 2 (*aux)->bf16, 3 (+resid)->f32
// ---------------------------------------------------------------------------
enum { EPI_BF16 = 0, EPI_SILU = 1, EPI_MUL = 2, EPI_RESID = 3 };

template<int EPI>
__global__ __launch_bounds__(256)
void gemm_bf16(const u16* __restrict__ A, const u16* __restrict__ W,
               const u16* __restrict__ aux, const float* __restrict__ resid,
               u16* __restrict__ outb, float* __restrict__ outf,
               int N, int K) {
  constexpr int BK  = 64;
  constexpr int LDT = BK + 8;                // 72 halves (TDM pad), rows 16B aligned
  __shared__ u16 As[2][128 * LDT];
  __shared__ u16 Ws[2][128 * LDT];

  const int tid  = threadIdx.x;
  const int wid  = tid >> 5;
  const int lane = tid & 31;
  const int l15  = lane & 15;
  const int lhi  = lane >> 4;
  const int wm   = (wid >> 2) * 64;
  const int wn   = (wid & 3) * 32;
  const long bM  = (long)blockIdx.y * 128;
  const long bN  = (long)blockIdx.x * 128;
  const int lr = tid >> 3;                   // 0..31 (prefetch row)
  const int lc = tid & 7;                    // 0..7  (prefetch chunk)

  f32x8 acc[4][2] = {};

  // prologue: DMA tile 0 into buffer 0
  if (wid == 0) {
    tdm_load_tile_2d(&A[bM * (long)K], lds_addr32(&As[0][0]), K, 128, BK);
    tdm_load_tile_2d(&W[bN * (long)K], lds_addr32(&Ws[0][0]), K, 128, BK);
  }

  int cur = 0;
  for (int k0 = 0; k0 < K; k0 += BK) {
    const bool has_next = (k0 + BK) < K;
    if (wid == 0) {
      if (has_next) {
        tdm_load_tile_2d(&A[bM * (long)K + k0 + BK], lds_addr32(&As[cur ^ 1][0]), K, 128, BK);
        tdm_load_tile_2d(&W[bN * (long)K + k0 + BK], lds_addr32(&Ws[cur ^ 1][0]), K, 128, BK);
        __builtin_amdgcn_s_wait_tensorcnt(2);   // current tile's 2 DMAs done
      } else {
        __builtin_amdgcn_s_wait_tensorcnt(0);
      }
    }
    __syncthreads();                            // current buffer valid for all waves
    if (has_next) {
      __builtin_prefetch(&A[(bM + lr) * (long)K + k0 + BK + lc * 8], 0, 1);
      __builtin_prefetch(&W[(bN + lr) * (long)K + k0 + BK + lc * 8], 0, 1);
    }
#pragma unroll
    for (int kk = 0; kk < BK; kk += 32) {
      FragBF a[4], b[2];
#pragma unroll
      for (int i = 0; i < 4; ++i) {
        const u16* p = &As[cur][(wm + i * 16 + l15) * LDT + kk + lhi * 8];
        a[i].u[0] = *(const u32x4*)p;
        a[i].u[1] = *(const u32x4*)(p + 16);
      }
#pragma unroll
      for (int j = 0; j < 2; ++j) {
        const u16* p = &Ws[cur][(wn + j * 16 + l15) * LDT + kk + lhi * 8];
        b[j].u[0] = *(const u32x4*)p;
        b[j].u[1] = *(const u32x4*)(p + 16);
      }
#pragma unroll
      for (int i = 0; i < 4; ++i)
#pragma unroll
        for (int j = 0; j < 2; ++j)
          acc[i][j] = __builtin_amdgcn_wmma_f32_16x16x32_bf16(
              false, a[i].v, false, b[j].v, (short)0, acc[i][j], false, false);
    }
    __syncthreads();                            // reads done before buffer reuse
    cur ^= 1;
  }

#pragma unroll
  for (int i = 0; i < 4; ++i)
#pragma unroll
    for (int j = 0; j < 2; ++j) {
      const long n0 = bN + wn + j * 16 + l15;
#pragma unroll
      for (int r = 0; r < 8; ++r) {
        const long row = bM + wm + i * 16 + r + lhi * 8;
        const long idx = row * (long)N + n0;
        float v = acc[i][j][r];
        if (EPI == EPI_BF16) {
          outb[idx] = f2bf(v);
        } else if (EPI == EPI_SILU) {
          v = v / (1.0f + __expf(-v));
          outb[idx] = f2bf(v);
        } else if (EPI == EPI_MUL) {
          v *= bf2f(aux[idx]);
          outb[idx] = f2bf(v);
        } else {
          outf[idx] = v + resid[idx];
        }
      }
    }
}

// ---------------------------------------------------------------------------
// Flash attention: grid (S/64 q-tiles, H, B), 128 threads = 4 waves.
// K tile staged once per block via async global->LDS b128 (ASYNCcnt); V tile
// transposed into LDS; per k-tile: QK^T (8 WMMA), online softmax with
// ds_swizzle row reductions, PV (8 WMMA). Causal mask applied analytically.
// ---------------------------------------------------------------------------
__global__ __launch_bounds__(128)
void flash_attn(const u16* __restrict__ q, const u16* __restrict__ k,
                const u16* __restrict__ v, u16* __restrict__ out) {
  constexpr int LDT = 72;
  __shared__ u16 Ks[64 * LDT];           // [kpos][d] K tile (async DMA)
  __shared__ u16 Vt[64 * LDT];           // [d][kpos] transposed V tile
  __shared__ u16 Ps[4 * 16 * LDT];       // per-wave 16x64 P tile

  const int tid  = threadIdx.x;
  const int wid  = tid >> 5;
  const int lane = tid & 31;
  const int l15  = lane & 15;
  const int lhi  = lane >> 4;
  const int qt   = blockIdx.x;
  const int head = blockIdx.y;
  const int bz   = blockIdx.z;
  const long base = ((long)bz * S_) * D_ + (long)head * HD_;
  const int qrow0 = qt * 64 + wid * 16;

  FragBF qf[2];                           // Q rows reused for all k-tiles
  {
    const u16* qp = &q[base + (long)(qrow0 + l15) * D_];
#pragma unroll
    for (int s = 0; s < 2; ++s) {
      const u16* p = qp + s * 32 + lhi * 8;
      qf[s].u[0] = *(const u32x4*)p;
      qf[s].u[1] = *(const u32x4*)(p + 16);
    }
  }

  f32x8 o[4] = {};
  float mrow[8], lrow[8];
#pragma unroll
  for (int r = 0; r < 8; ++r) { mrow[r] = -__builtin_inff(); lrow[r] = 0.0f; }

  u16* ps = &Ps[wid * 16 * LDT];

  for (int kt = 0; kt <= qt; ++kt) {
    __syncthreads();
    // async-DMA the K tile into LDS (row-major, padded stride)
#pragma unroll
    for (int it = 0; it < 4; ++it) {
      const int idx = tid + it * 128;     // 0..511
      const int kr  = idx >> 3;           // key row 0..63
      const int c   = idx & 7;            // 16B chunk 0..7
      async_load_b128(&k[base + (long)(kt * 64 + kr) * D_ + c * 8],
                      lds_addr32(&Ks[kr * LDT + c * 8]));
    }
    // stage V tile transposed into LDS: Vt[d][kpos]
#pragma unroll
    for (int it = 0; it < 4; ++it) {
      const int idx = tid + it * 128;
      const int kr  = idx >> 3;
      const int c   = idx & 7;
      Pack8 t;
      t.u = *(const u32x4*)&v[base + (long)(kt * 64 + kr) * D_ + c * 8];
#pragma unroll
      for (int e = 0; e < 8; ++e)
        Vt[(c * 8 + e) * LDT + kr] = t.h[e];
    }
    wait_asynccnt0();
    __syncthreads();

    // S = Q * K^T (K rows from LDS load directly as B fragments)
    f32x8 sc[4] = {};
#pragma unroll
    for (int s = 0; s < 2; ++s) {
      FragBF kf[4];
#pragma unroll
      for (int j = 0; j < 4; ++j) {
        const u16* p = &Ks[(j * 16 + l15) * LDT + s * 32 + lhi * 8];
        kf[j].u[0] = *(const u32x4*)p;
        kf[j].u[1] = *(const u32x4*)(p + 16);
      }
#pragma unroll
      for (int j = 0; j < 4; ++j)
        sc[j] = __builtin_amdgcn_wmma_f32_16x16x32_bf16(
            false, qf[s].v, false, kf[j].v, (short)0, sc[j], false, false);
    }

    // online softmax (rows live at fixed VGPR r across the 16-lane half)
#pragma unroll
    for (int r = 0; r < 8; ++r) {
      const int qrow = qrow0 + r + lhi * 8;
      float mx = -__builtin_inff();
#pragma unroll
      for (int j = 0; j < 4; ++j) {
        const int col = kt * 64 + j * 16 + l15;
        float val = sc[j][r] * 0.125f;                 // 1/sqrt(64)
        val = (col <= qrow) ? val : -__builtin_inff(); // causal
        sc[j][r] = val;
        mx = fmaxf(mx, val);
      }
      mx = red16_max(mx);
      const float mnew = fmaxf(mrow[r], mx);
      const float corr = __expf(mrow[r] - mnew);
      float psum = 0.0f;
#pragma unroll
      for (int j = 0; j < 4; ++j) {
        const float p = __expf(sc[j][r] - mnew);
        sc[j][r] = p;
        psum += p;
      }
      psum = red16_add(psum);
      lrow[r] = lrow[r] * corr + psum;
      mrow[r] = mnew;
#pragma unroll
      for (int j = 0; j < 4; ++j) o[j][r] = o[j][r] * corr;
    }

    // P -> LDS (per-wave tile; same-wave LDS ops are in order)
#pragma unroll
    for (int j = 0; j < 4; ++j)
#pragma unroll
      for (int r = 0; r < 8; ++r)
        ps[(r + lhi * 8) * LDT + j * 16 + l15] = f2bf(sc[j][r]);

    // O += P * V
#pragma unroll
    for (int s = 0; s < 2; ++s) {
      FragBF pf, vf[4];
      const u16* pp = &ps[l15 * LDT + s * 32 + lhi * 8];
      pf.u[0] = *(const u32x4*)pp;
      pf.u[1] = *(const u32x4*)(pp + 16);
#pragma unroll
      for (int j = 0; j < 4; ++j) {
        const u16* p = &Vt[(j * 16 + l15) * LDT + s * 32 + lhi * 8];
        vf[j].u[0] = *(const u32x4*)p;
        vf[j].u[1] = *(const u32x4*)(p + 16);
      }
#pragma unroll
      for (int j = 0; j < 4; ++j)
        o[j] = __builtin_amdgcn_wmma_f32_16x16x32_bf16(
            false, pf.v, false, vf[j].v, (short)0, o[j], false, false);
    }
  }

  // normalize, store bf16
#pragma unroll
  for (int j = 0; j < 4; ++j)
#pragma unroll
    for (int r = 0; r < 8; ++r) {
      const float val = o[j][r] / lrow[r];
      const long row = (long)bz * S_ + qrow0 + r + lhi * 8;
      out[row * (long)D_ + head * HD_ + j * 16 + l15] = f2bf(val);
    }
}

// ---------------------------------------------------------------------------
// launch
// ---------------------------------------------------------------------------
extern "C" void kernel_launch(void* const* d_in, const int* in_sizes, int n_in,
                              void* d_out, int out_size, void* d_ws, size_t ws_size,
                              hipStream_t stream) {
  (void)in_sizes; (void)n_in; (void)out_size; (void)ws_size;
  const float* x  = (const float*)d_in[0];
  // d_in[1] = attention_mask (causal, applied analytically)
  const float* wq = (const float*)d_in[2];
  const float* wk = (const float*)d_in[3];
  const float* wv = (const float*)d_in[4];
  const float* wo = (const float*)d_in[5];
  const float* w1 = (const float*)d_in[6];
  const float* w2 = (const float*)d_in[7];
  const float* w3 = (const float*)d_in[8];
  const float* ga = (const float*)d_in[9];
  const float* gf = (const float*)d_in[10];
  float* out = (float*)d_out;

  char* ws = (char*)d_ws;
  size_t off = 0;
  auto alloc = [&](size_t bytes) -> void* {
    void* p = ws + off;
    off = (off + bytes + 255) & ~(size_t)255;
    return p;
  };
  const size_t DD = (size_t)D_ * D_;
  const size_t FD = (size_t)DFF_ * D_;
  const size_t MD = (size_t)M_ * D_;
  const size_t MF = (size_t)M_ * DFF_;

  u16* wqb = (u16*)alloc(DD * 2);
  u16* wkb = (u16*)alloc(DD * 2);
  u16* wvb = (u16*)alloc(DD * 2);
  u16* wob = (u16*)alloc(DD * 2);
  u16* w1b = (u16*)alloc(FD * 2);
  u16* w2b = (u16*)alloc(FD * 2);
  u16* w3b = (u16*)alloc(FD * 2);
  u16* nrm = (u16*)alloc(MD * 2);
  u16* qb  = (u16*)alloc(MD * 2);
  u16* kb  = (u16*)alloc(MD * 2);
  u16* vb  = (u16*)alloc(MD * 2);
  u16* ab  = (u16*)alloc(MD * 2);
  float* x1 = (float*)alloc(MD * 4);
  u16* su  = (u16*)alloc(MF * 2);
  u16* gt  = (u16*)alloc(MF * 2);

  auto cvt = [&](const float* s, u16* d, size_t n) {
    cvt_f32_bf16<<<(unsigned)((n + 255) / 256), 256, 0, stream>>>(s, d, (long)n);
  };
  cvt(wq, wqb, DD); cvt(wk, wkb, DD); cvt(wv, wvb, DD); cvt(wo, wob, DD);
  cvt(w1, w1b, FD); cvt(w2, w2b, FD); cvt(w3, w3b, FD);

  rmsnorm_bf16<<<M_, 256, 0, stream>>>(x, ga, nrm);

  dim3 g1(D_ / 128, M_ / 128);      // (8, 32)
  gemm_bf16<EPI_BF16><<<g1, 256, 0, stream>>>(nrm, wqb, nullptr, nullptr, qb, nullptr, D_, D_);
  gemm_bf16<EPI_BF16><<<g1, 256, 0, stream>>>(nrm, wkb, nullptr, nullptr, kb, nullptr, D_, D_);
  gemm_bf16<EPI_BF16><<<g1, 256, 0, stream>>>(nrm, wvb, nullptr, nullptr, vb, nullptr, D_, D_);

  const long nrope = (long)M_ * H_ * 32;
  rope_inplace<<<(unsigned)((nrope + 255) / 256), 256, 0, stream>>>(qb);
  rope_inplace<<<(unsigned)((nrope + 255) / 256), 256, 0, stream>>>(kb);

  flash_attn<<<dim3(S_ / 64, H_, B_), 128, 0, stream>>>(qb, kb, vb, ab);

  // x1 = x + attn @ wo^T
  gemm_bf16<EPI_RESID><<<g1, 256, 0, stream>>>(ab, wob, nullptr, x, nullptr, x1, D_, D_);

  rmsnorm_bf16<<<M_, 256, 0, stream>>>(x1, gf, nrm);

  dim3 g2(DFF_ / 128, M_ / 128);    // (32, 32)
  gemm_bf16<EPI_SILU><<<g2, 256, 0, stream>>>(nrm, w1b, nullptr, nullptr, su, nullptr, DFF_, D_);
  gemm_bf16<EPI_MUL ><<<g2, 256, 0, stream>>>(nrm, w3b, su, nullptr, gt, nullptr, DFF_, D_);

  // out = x1 + gate @ w2^T
  gemm_bf16<EPI_RESID><<<g1, 256, 0, stream>>>(gt, w2b, nullptr, x1, nullptr, out, D_, DFF_);
}
